// Router_34213709480498
// MI455X (gfx1250) — compile-verified, matrix operand order checked by reading, not run on hardware
//
#include <hip/hip_runtime.h>
#include <hip/hip_bf16.h>
#include <math.h>
#include <stdint.h>

// ---------------- problem constants ----------------
#define B_    4
#define S_    1024
#define D_    1024
#define DS_   64
#define R_    (B_ * S_)     // 4096 rows
#define NQK   4096
#define NV    4096
#define NKN   8192
#define KQK   32
#define KV    64
#define KKN   64
#define KEEP  0.9f

typedef __attribute__((ext_vector_type(16))) __bf16 v16bf;
typedef __attribute__((ext_vector_type(8)))  float  v8f;
typedef __attribute__((ext_vector_type(4)))  unsigned int u32x4;
typedef __attribute__((ext_vector_type(8)))  int    i32x8;
typedef __attribute__((ext_vector_type(4)))  int    i32x4;

// 16-bit A-matrix 16x32 lane layout (ISA 7.12.2):
// lane 0-15 hold K = {0..7, 16..23}, lanes 16-31 hold K = {8..15, 24..31},
// VGPR v holds the pair {K0, K0+1}.
__device__ __forceinline__ int wmma_k0(int v, int hi) {
    int base = (v < 4) ? (2 * v) : (16 + 2 * (v - 4));
    return base + (hi ? 8 : 0);
}

// ---------------- Tensor Data Mover: 2D tile global -> LDS ----------------
// Builds a D# per CDNA5 ISA ch.8: group0 = {count/type/addr}, group1 = dims.
// data_size_code: 0=1B, 1=2B, 2=4B. tile_d0/tensor_d0 in elements, stride in elements.
__device__ __forceinline__ void tdm_load_2d(void* lds_dst, const void* gsrc,
                                            unsigned data_size_code,
                                            unsigned tile_d0, unsigned tile_d1,
                                            unsigned long long stride_d0) {
    unsigned lds_addr = (unsigned)(uintptr_t)lds_dst;      // LDS aperture offset = addr[31:0]
    unsigned long long ga = (unsigned long long)(uintptr_t)gsrc;
    u32x4 g0;
    g0[0] = 1u;                                            // count=1, user descriptor
    g0[1] = lds_addr;                                      // lds_addr (bits 63:32)
    g0[2] = (unsigned)(ga & 0xFFFFFFFFu);                  // global_addr[31:0]
    g0[3] = (unsigned)((ga >> 32) & 0x01FFFFFFu) | 0x80000000u; // addr[56:32] | type=2
    unsigned long long d0 = tile_d0;                       // tensor_dim0 = tile width
    unsigned long long d1 = tile_d1;                       // tensor_dim1 = tile height
    i32x8 g1;
    g1[0] = (int)(data_size_code << 16);                   // mask=0, data_size, no flags
    g1[1] = (int)((d0 & 0xFFFFu) << 16);                   // tensor_dim0[15:0]
    g1[2] = (int)(((d0 >> 16) & 0xFFFFu) | ((d1 & 0xFFFFu) << 16));
    g1[3] = (int)(((d1 >> 16) & 0xFFFFu) | ((unsigned long long)tile_d0 << 16)); // tile_dim0
    g1[4] = (int)(tile_d1 & 0xFFFFu);                      // tile_dim1, tile_dim2=0
    g1[5] = (int)(stride_d0 & 0xFFFFFFFFull);              // tensor_dim0_stride[31:0]
    g1[6] = (int)((stride_d0 >> 32) & 0xFFFFull);          // tensor_dim0_stride[47:32]
    g1[7] = 0;
    i32x4 gz = {0, 0, 0, 0};
#if __has_include(<hip/amd_detail/amd_gfx1250_TDM.h>)
    i32x8 gz8 = {0, 0, 0, 0, 0, 0, 0, 0};
    __builtin_amdgcn_tensor_load_to_lds(g0, g1, gz, gz, gz8, 0);   // 6-arg toolchain
#else
    __builtin_amdgcn_tensor_load_to_lds(g0, g1, gz, gz, 0);        // 5-arg toolchain
#endif
}

// ---------------- kernel 0: init accumulators + aux outputs ----------------
__global__ void init_kernel(float* __restrict__ acc, int n,
                            float* __restrict__ aux_attn, float* __restrict__ aux_know) {
    int i = blockIdx.x * blockDim.x + threadIdx.x;
    if (i < n) acc[i] = 0.0f;
    if (i == 0) { *aux_attn = 0.0f; *aux_know = 0.0f; }
}

// ---------------- kernel 1: normalize neuron_emb -> bf16 ----------------
__global__ void norm_emb_kernel(const float* __restrict__ emb, __bf16* __restrict__ embn) {
    int wave = threadIdx.x >> 5, lane = threadIdx.x & 31;
    int row = blockIdx.x * 8 + wave;
    const float* er = emb + (size_t)row * DS_;
    float v0 = er[lane], v1 = er[lane + 32];
    float ss = v0 * v0 + v1 * v1;
    #pragma unroll
    for (int off = 16; off > 0; off >>= 1) ss += __shfl_xor(ss, off, 32);
    float inv = 1.0f / (sqrtf(ss) + 1e-8f);
    __bf16* orow = embn + (size_t)row * DS_;
    orow[lane]      = (__bf16)(v0 * inv);
    orow[lane + 32] = (__bf16)(v1 * inv);
}

// ---------------- kernel 2: h = (x @ W + b) / keep  -> bf16 ----------------
// grid.x = R/16 row tiles; one wave per 16-col tile; 64-wide K panels, 2 WMMA/iter
__global__ void gemm_h_kernel(const float* __restrict__ x, const float* __restrict__ W,
                              const float* __restrict__ bias, int C,
                              __bf16* __restrict__ out) {
    __shared__ __bf16 As[16][64];
    const int lane = threadIdx.x & 31;
    const int wave = threadIdx.x >> 5;
    const int row0 = blockIdx.x * 16;
    const int n0   = wave * 16;
    const int hi   = lane >> 4;
    const int nl   = lane & 15;
    v8f acc = {};
    for (int kk = 0; kk < D_; kk += 64) {
        for (int i = threadIdx.x; i < 16 * 64; i += blockDim.x) {
            int mm = i >> 6, k = i & 63;
            As[mm][k] = (__bf16)x[(size_t)(row0 + mm) * D_ + kk + k];
        }
        __syncthreads();
        v16bf a0, a1, b0, b1;
        #pragma unroll
        for (int v = 0; v < 8; ++v) {
            int k0 = wmma_k0(v, hi);
            a0[2 * v]     = As[nl][k0];
            a0[2 * v + 1] = As[nl][k0 + 1];
            a1[2 * v]     = As[nl][32 + k0];
            a1[2 * v + 1] = As[nl][32 + k0 + 1];
            b0[2 * v]     = (__bf16)W[(size_t)(kk + k0)          * C + n0 + nl];
            b0[2 * v + 1] = (__bf16)W[(size_t)(kk + k0 + 1)      * C + n0 + nl];
            b1[2 * v]     = (__bf16)W[(size_t)(kk + 32 + k0)     * C + n0 + nl];
            b1[2 * v + 1] = (__bf16)W[(size_t)(kk + 32 + k0 + 1) * C + n0 + nl];
        }
        acc = __builtin_amdgcn_wmma_f32_16x16x32_bf16(false, a0, false, b0,
                                                      (short)0, acc, false, false);
        acc = __builtin_amdgcn_wmma_f32_16x16x32_bf16(false, a1, false, b1,
                                                      (short)0, acc, false, false);
        __syncthreads();
    }
    const float inv_keep = 1.0f / KEEP;
    #pragma unroll
    for (int r = 0; r < 8; ++r) {
        int row = row0 + r + 8 * hi;
        int col = n0 + nl;
        out[(size_t)row * C + col] = (__bf16)((acc[r] + bias[col]) * inv_keep);
    }
}

// ---------------- kernel 3: tau = x @ W_tau + b_tau ----------------
__global__ void tau_kernel(const float* __restrict__ x,
                           const float* __restrict__ Wta, const float* __restrict__ bta,
                           const float* __restrict__ Wtk, const float* __restrict__ btk,
                           float* __restrict__ tau) {
    int wave = threadIdx.x >> 5, lane = threadIdx.x & 31;
    int row = blockIdx.x * 8 + wave;
    const float* xr = x + (size_t)row * D_;
    float s0 = 0, s1 = 0, s2 = 0, s3 = 0;
    for (int k = lane; k < D_; k += 32) {
        float xv = xr[k];
        s0 += xv * Wta[k * 3 + 0];
        s1 += xv * Wta[k * 3 + 1];
        s2 += xv * Wta[k * 3 + 2];
        s3 += xv * Wtk[k];
    }
    #pragma unroll
    for (int off = 16; off > 0; off >>= 1) {
        s0 += __shfl_xor(s0, off, 32);
        s1 += __shfl_xor(s1, off, 32);
        s2 += __shfl_xor(s2, off, 32);
        s3 += __shfl_xor(s3, off, 32);
    }
    if (lane == 0) {
        tau[row * 4 + 0] = s0 + bta[0];
        tau[row * 4 + 1] = s1 + bta[1];
        tau[row * 4 + 2] = s2 + bta[2];
        tau[row * 4 + 3] = s3 + btk[0];
    }
}

// ---------------- kernel 4: scores + soft-threshold gate (WMMA, K=64) ----------------
// block = 128 (4 waves); each wave owns 4 consecutive 16x16 n-tiles (8 WMMAs,
// A fragments reused). A tile fetched via TDM. grid = (R/16, Ntiles/16)
__global__ void scores_gate_kernel(const __bf16* __restrict__ h, int hC, int hcol0,
                                   const __bf16* __restrict__ emb,
                                   const float* __restrict__ tau, int taucol,
                                   float* __restrict__ e_out, int N) {
    __shared__ __bf16 As[16][64];
    const int lane = threadIdx.x & 31;
    const int wave = threadIdx.x >> 5;
    const int row0 = blockIdx.x * 16;
    if (threadIdx.x == 0) {
        // 16 rows x 64 bf16, row stride hC elements: strided 2D tile -> TDM
        tdm_load_2d(&As[0][0], h + (size_t)row0 * hC + hcol0,
                    /*2B*/1, /*d0*/64, /*d1*/16, (unsigned long long)hC);
        __builtin_amdgcn_s_wait_tensorcnt(0);
    }
    __syncthreads();
    const int hi = lane >> 4;
    const int nl = lane & 15;
    v16bf a0, a1;
    #pragma unroll
    for (int v = 0; v < 8; ++v) {
        int k0 = wmma_k0(v, hi);
        a0[2 * v]     = As[nl][k0];
        a0[2 * v + 1] = As[nl][k0 + 1];
        a1[2 * v]     = As[nl][32 + k0];
        a1[2 * v + 1] = As[nl][32 + k0 + 1];
    }
    float tsub[8];
    #pragma unroll
    for (int r = 0; r < 8; ++r)
        tsub[r] = tau[(row0 + r + 8 * hi) * 4 + taucol];
    #pragma unroll
    for (int t = 0; t < 4; ++t) {
        int n0 = (blockIdx.y * 16 + wave * 4 + t) * 16;
        const __bf16* brow = emb + (size_t)(n0 + nl) * DS_;
        __builtin_prefetch(brow + DS_, 0, 0);   // global_prefetch_b8 (next emb row)
        v16bf b0, b1;
        #pragma unroll
        for (int v = 0; v < 8; ++v) {
            int k0 = wmma_k0(v, hi);
            b0[2 * v]     = brow[k0];
            b0[2 * v + 1] = brow[k0 + 1];
            b1[2 * v]     = brow[32 + k0];
            b1[2 * v + 1] = brow[32 + k0 + 1];
        }
        v8f acc = {};
        acc = __builtin_amdgcn_wmma_f32_16x16x32_bf16(false, a0, false, b0,
                                                      (short)0, acc, false, false);
        acc = __builtin_amdgcn_wmma_f32_16x16x32_bf16(false, a1, false, b1,
                                                      (short)0, acc, false, false);
        #pragma unroll
        for (int r = 0; r < 8; ++r) {
            int row  = row0 + r + 8 * hi;
            int col  = n0 + nl;
            float raw  = acc[r] - tsub[r];
            float gate = raw > 0.0f ? raw : 1e-8f * __expf(raw);
            float e    = __expf(gate) - 1.0f;
            e_out[(size_t)row * N + col] = e;
        }
    }
}

// ---------------- kernel 5: per-row top-k selection + normalization ----------------
// one 256-thread block per row; row pulled into LDS by the Tensor Data Mover,
// then iterative argmax (emerges sorted descending, matching lax.top_k)
__global__ void topk_kernel(const float* __restrict__ e, int N, int K,
                            float* __restrict__ out_g, float* __restrict__ out_i,
                            float* __restrict__ acc) {
    __shared__ float sbuf[8192];
    __shared__ float wbest[8];
    __shared__ int   wbidx[8];
    __shared__ float selv[64];
    __shared__ int   seli[64];
    __shared__ float ssum, sstr;
    const int row = blockIdx.x;
    const int tid = threadIdx.x, lane = tid & 31, wave = tid >> 5;
    const float* er = e + (size_t)row * N;
    if (tid == 0) {
        tdm_load_2d(sbuf, er, /*4B*/2, (unsigned)N, 1, (unsigned long long)N);
        __builtin_amdgcn_s_wait_tensorcnt(0);
    }
    __syncthreads();
    for (int j = 0; j < K; ++j) {
        float bv = -1e30f; int bi = 0;
        for (int i = tid; i < N; i += blockDim.x) {
            float v = sbuf[i];
            if (v > bv) { bv = v; bi = i; }
        }
        #pragma unroll
        for (int off = 16; off > 0; off >>= 1) {
            float ov = __shfl_xor(bv, off, 32);
            int   oi = __shfl_xor(bi, off, 32);
            if (ov > bv) { bv = ov; bi = oi; }
        }
        if (lane == 0) { wbest[wave] = bv; wbidx[wave] = bi; }
        __syncthreads();
        if (tid == 0) {
            float fv = wbest[0]; int fi = wbidx[0];
            #pragma unroll
            for (int w = 1; w < 8; ++w)
                if (wbest[w] > fv) { fv = wbest[w]; fi = wbidx[w]; }
            selv[j] = fv; seli[j] = fi;
            sbuf[fi] = -1e30f;
        }
        __syncthreads();
    }
    if (tid == 0) {
        float s = 1e-8f;
        for (int j = 0; j < K; ++j) s += selv[j];
        ssum = s;
        sstr = tanhf(selv[0]);   // gate_strength = tanh(max exp_gate)
    }
    __syncthreads();
    if (tid < K) {
        float g = selv[tid] / ssum * sstr;
        out_g[(size_t)row * K + tid] = g;
        out_i[(size_t)row * K + tid] = (float)seli[tid];
        atomicAdd(&acc[seli[tid]], g);
    }
}

// ---------------- kernel 6: aux losses from per-neuron sums ----------------
__global__ void aux_kernel(const float* __restrict__ acc,
                           float* __restrict__ aux_attn, float* __restrict__ aux_know) {
    int i = blockIdx.x * blockDim.x + threadIdx.x;   // 0..20479
    bool know   = (i >= 3 * NQK);                     // split at 12288 (block-uniform)
    float t     = know ? (1.0f / (float)NKN) : (1.0f / (float)NQK);
    float scale = know ? (float)NKN : (float)NQK;
    float d = acc[i] * (1.0f / (float)R_) - t;
    float c = d * d * scale;
    #pragma unroll
    for (int off = 16; off > 0; off >>= 1) c += __shfl_xor(c, off, 32);
    if ((threadIdx.x & 31) == 0) atomicAdd(know ? aux_know : aux_attn, c);
}

// ---------------- host launch ----------------
extern "C" void kernel_launch(void* const* d_in, const int* in_sizes, int n_in,
                              void* d_out, int out_size, void* d_ws, size_t ws_size,
                              hipStream_t stream) {
    (void)in_sizes; (void)n_in; (void)out_size; (void)ws_size;
    const float* x      = (const float*)d_in[0];
    const float* emb    = (const float*)d_in[1];
    const float* W_attn = (const float*)d_in[2];
    const float* b_attn = (const float*)d_in[3];
    const float* W_know = (const float*)d_in[4];
    const float* b_know = (const float*)d_in[5];
    const float* W_ta   = (const float*)d_in[6];
    const float* b_ta   = (const float*)d_in[7];
    const float* W_tk   = (const float*)d_in[8];
    const float* b_tk   = (const float*)d_in[9];
    float* out = (float*)d_out;

    // workspace layout
    char* ws = (char*)d_ws;
    size_t off = 0;
    __bf16* embn = (__bf16*)(ws + off); off += (size_t)(NQK + NV + NKN) * DS_ * sizeof(__bf16);
    __bf16* hA   = (__bf16*)(ws + off); off += (size_t)R_ * 192 * sizeof(__bf16);
    __bf16* hK   = (__bf16*)(ws + off); off += (size_t)R_ * DS_ * sizeof(__bf16);
    float*  tau  = (float*) (ws + off); off += (size_t)R_ * 4 * sizeof(float);
    float*  acc  = (float*) (ws + off); off += (size_t)(3 * NQK + NKN) * sizeof(float);
    float*  eQ   = (float*) (ws + off); off += (size_t)R_ * NQK * sizeof(float);
    float*  eK   = (float*) (ws + off); off += (size_t)R_ * NQK * sizeof(float);
    float*  eV   = (float*) (ws + off); off += (size_t)R_ * NV  * sizeof(float);
    float*  eKN  = (float*) (ws + off); off += (size_t)R_ * NKN * sizeof(float);

    // output offsets (flat tuple, all f32; indices stored as float)
    float* gQ = out;                 float* iQ = out + (size_t)R_ * KQK;
    float* gK = iQ + (size_t)R_*KQK; float* iK = gK + (size_t)R_ * KQK;
    float* gV = iK + (size_t)R_*KQK; float* iV = gV + (size_t)R_ * KV;
    float* aux_attn = iV + (size_t)R_ * KV;
    float* gkn = aux_attn + 1;       float* ikn = gkn + (size_t)R_ * KKN;
    float* aux_know = ikn + (size_t)R_ * KKN;

    const int NACC = 3 * NQK + NKN;  // 20480
    init_kernel<<<(NACC + 255) / 256, 256, 0, stream>>>(acc, NACC, aux_attn, aux_know);

    norm_emb_kernel<<<(NQK + NV + NKN) / 8, 256, 0, stream>>>(emb, embn);

    gemm_h_kernel<<<R_ / 16, 384, 0, stream>>>(x, W_attn, b_attn, 192, hA);  // 12 waves
    gemm_h_kernel<<<R_ / 16, 128, 0, stream>>>(x, W_know, b_know, DS_, hK);  // 4 waves

    tau_kernel<<<R_ / 8, 256, 0, stream>>>(x, W_ta, b_ta, W_tk, b_tk, tau);

    const __bf16* v_emb  = embn + (size_t)NQK * DS_;
    const __bf16* kn_emb = embn + (size_t)(NQK + NV) * DS_;
    dim3 gAttn(R_ / 16, NQK / 16 / 16);   // (256, 16): 4 waves x 4 tiles each
    dim3 gKnow(R_ / 16, NKN / 16 / 16);   // (256, 32)
    scores_gate_kernel<<<gAttn, 128, 0, stream>>>(hA, 192,   0, embn,   tau, 0, eQ,  NQK);
    scores_gate_kernel<<<gAttn, 128, 0, stream>>>(hA, 192,  64, embn,   tau, 1, eK,  NQK);
    scores_gate_kernel<<<gAttn, 128, 0, stream>>>(hA, 192, 128, v_emb,  tau, 2, eV,  NV);
    scores_gate_kernel<<<gKnow, 128, 0, stream>>>(hK, DS_,   0, kn_emb, tau, 3, eKN, NKN);

    topk_kernel<<<R_, 256, 0, stream>>>(eQ,  NQK, KQK, gQ,  iQ,  acc);
    topk_kernel<<<R_, 256, 0, stream>>>(eK,  NQK, KQK, gK,  iK,  acc + NQK);
    topk_kernel<<<R_, 256, 0, stream>>>(eV,  NV,  KV,  gV,  iV,  acc + 2 * NQK);
    topk_kernel<<<R_, 256, 0, stream>>>(eKN, NKN, KKN, gkn, ikn, acc + 3 * NQK);

    aux_kernel<<<NACC / 256, 256, 0, stream>>>(acc, aux_attn, aux_know);
}
